// CenterFaceMask_40346922779296
// MI455X (gfx1250) — matrix operand order repaired and checked by reading, not compiled, not run on hardware
//
#include <hip/hip_runtime.h>
#include <hip/hip_bf16.h>

// ---- problem constants (match reference) ----
constexpr int kH    = 512;
constexpr int kW    = 512;
constexpr int kHW   = kH * kW;        // 262144
constexpr int kNORG = 10;
constexpr int kSS   = 32;
constexpr int kNBLK = 128;            // argmax partial blocks per organ

typedef __attribute__((ext_vector_type(16))) _Float16 v16h;
typedef __attribute__((ext_vector_type(8)))  float    v8f;

__device__ __forceinline__ float sigmoidf_(float x) {
    return 1.0f / (1.0f + __expf(-x));
}

// ---------------------------------------------------------------------------
// K1: per-organ partial argmax over heat channel (first-max tie-break)
// ---------------------------------------------------------------------------
__global__ void k_argmax_partial(const float* __restrict__ feat,
                                 float* __restrict__ pval,
                                 int*   __restrict__ pidx) {
    const int o   = blockIdx.y;
    const int blk = blockIdx.x;
    const int tid = threadIdx.x;
    const float* heat = feat + (size_t)(1027 + o) * kHW;
    const int chunk = kHW / kNBLK;                 // 2048
    const int base  = blk * chunk;

    float bv = -3.4e38f;
    int   bi = 0x7fffffff;
    for (int k = tid; k < chunk; k += 256) {
        const int i = base + k;
        __builtin_prefetch(heat + i + 512, 0, 1);  // global_prefetch_b8
        const float v = heat[i];
        if (v > bv || (v == bv && i < bi)) { bv = v; bi = i; }
    }

    __shared__ float sv[256];
    __shared__ int   si[256];
    sv[tid] = bv; si[tid] = bi;
    __syncthreads();
    for (int s = 128; s > 0; s >>= 1) {
        if (tid < s) {
            const float v = sv[tid + s];
            const int   i = si[tid + s];
            if (v > sv[tid] || (v == sv[tid] && i < si[tid])) { sv[tid] = v; si[tid] = i; }
        }
        __syncthreads();
    }
    if (tid == 0) {
        pval[o * kNBLK + blk] = sv[0];
        pidx[o * kNBLK + blk] = si[0];
    }
}

// ---------------------------------------------------------------------------
// K2: final argmax reduce + shape-vector gather + per-organ params + centers
// params[o*16 + {0..8}] = {idx, px, py, sh, sw, r0, c0, Rstart, Hc}
// ---------------------------------------------------------------------------
__global__ void k_argmax_gather(const float* __restrict__ feat,
                                const float* __restrict__ pval,
                                const int*   __restrict__ pidx,
                                int*   __restrict__ params,
                                float* __restrict__ shape,
                                float* __restrict__ out) {
    const int o   = blockIdx.x;
    const int tid = threadIdx.x;                   // 0..1023

    __shared__ float sv[kNBLK];
    __shared__ int   si[kNBLK];
    if (tid < kNBLK) { sv[tid] = pval[o * kNBLK + tid]; si[tid] = pidx[o * kNBLK + tid]; }
    __syncthreads();
    for (int s = kNBLK / 2; s > 0; s >>= 1) {
        if (tid < s) {
            const float v = sv[tid + s];
            const int   i = si[tid + s];
            if (v > sv[tid] || (v == sv[tid] && i < si[tid])) { sv[tid] = v; si[tid] = i; }
        }
        __syncthreads();
    }
    const int idx = si[0];

    // gather 1024-dim shape vector (channels 1..1024) at the peak
    shape[o * 1024 + tid] = feat[(size_t)(1 + tid) * kHW + idx];

    if (tid == 0) {
        const int px = idx % kW;
        const int py = idx / kW;
        int shv = (int)feat[(size_t)1025 * kHW + idx];   // trunc toward zero
        int swv = (int)feat[(size_t)1026 * kHW + idx];
        shv = shv < 0 ? -shv : shv;  if (shv < 1) shv = 1;
        swv = swv < 0 ? -swv : swv;  if (swv < 1) swv = 1;
        const int r0 = py - shv / 2;
        const int c0 = px - swv / 2;
        const int Rs = r0 > 0 ? r0 : 0;
        const int Re = (r0 + shv) < kH ? (r0 + shv) : kH;
        int* P = params + o * 16;
        P[0] = idx; P[1] = px; P[2] = py; P[3] = shv; P[4] = swv;
        P[5] = r0;  P[6] = c0; P[7] = Rs; P[8] = Re - Rs;
        // centers output (px, py) as float
        out[2 * kHW + o * 2 + 0] = (float)px;
        out[2 * kHW + o * 2 + 1] = (float)py;
    }
}

// ---------------------------------------------------------------------------
// K3: row-pass of separable bilinear via WMMA:  T(Hc x 32) = Wy(Hc x 32) * S(32 x 32)
// One wave32 per 16-row tile; one v_wmma_f32_16x16x32_f16 per 16-col half.
// ---------------------------------------------------------------------------
__global__ void k_rowinterp_wmma(const float* __restrict__ shape,
                                 const int*   __restrict__ params,
                                 float* __restrict__ T) {
    const int o    = blockIdx.y;
    const int tile = blockIdx.x;
    const int* P   = params + o * 16;
    const int Hc   = P[8];
    if (tile * 16 >= Hc) return;                   // wave-uniform -> EXEC stays all-1s

    const int Rs  = P[7];
    const int r0  = P[5];
    const int shv = P[3];

    const int lane = threadIdx.x;                  // 0..31 (wave32)
    const int m    = lane & 15;                    // A: lane -> M row
    const int kb   = (lane >> 4) << 3;             // A: lane half -> K base (0 or 8)

    const int r = tile * 16 + m;                   // local row in clipped box
    const int R = Rs + r;                          // image row
    float sy = ((float)(R - r0) + 0.5f) * (32.0f / (float)shv) - 0.5f;
    sy = fminf(fmaxf(sy, 0.0f), 31.0f);
    const int   y0 = (int)floorf(sy);
    const int   y1 = (y0 + 1) < 31 ? (y0 + 1) : 31;
    const float wy = sy - (float)y0;
    const float w0 = 1.0f - wy;
    const bool valid = (r < Hc);

    // A matrix (16x32 f16): element e -> K per ISA 16-bit A layout
    v16h a;
#pragma unroll
    for (int e = 0; e < 16; ++e) {
        const int K = (e < 8) ? (kb + e) : (16 + kb + (e - 8));
        float w = ((K == y0) ? w0 : 0.0f) + ((K == y1) ? wy : 0.0f);
        a[e] = (_Float16)(valid ? w : 0.0f);
    }

    // B matrix (32x16 f16): lane&15 = N (shape col x), element e -> K (shape row y) as A
    const float* srow = shape + o * 1024 + (lane & 15);
    v16h b0, b1;
#pragma unroll
    for (int e = 0; e < 16; ++e) {
        const int K = (e < 8) ? (kb + e) : (16 + kb + (e - 8));
        b0[e] = (_Float16)srow[K * 32];
        b1[e] = (_Float16)srow[K * 32 + 16];
    }

    v8f c = {};
    v8f d0 = __builtin_amdgcn_wmma_f32_16x16x32_f16(false, a, false, b0,
                                                    (short)0, c, false, false);
    v8f d1 = __builtin_amdgcn_wmma_f32_16x16x32_f16(false, a, false, b1,
                                                    (short)0, c, false, false);

    // D layout: VGPR j -> M = j (lanes 0-15) / j+8 (lanes 16-31), N = lane&15
    const int half = (lane >> 4) << 3;
    const int col  = lane & 15;
#pragma unroll
    for (int j = 0; j < 8; ++j) {
        const int row = tile * 16 + j + half;
        if (row < Hc) {
            T[(o * 512 + row) * 32 + col]      = d0[j];
            T[(o * 512 + row) * 32 + col + 16] = d1[j];
        }
    }
}

// ---------------------------------------------------------------------------
// K4: streaming final map: inside-box test, 2-tap x-interp from T, sigmoids
// ---------------------------------------------------------------------------
__global__ void k_final(const float* __restrict__ feat,
                        const int*   __restrict__ params,
                        const float* __restrict__ T,
                        float* __restrict__ out) {
    const int o = blockIdx.y;
    const int p = blockIdx.x * blockDim.x + threadIdx.x;
    const int R = p >> 9;                          // /512
    const int C = p & 511;
    const int* P = params + o * 16;
    const int shv = P[3], swv = P[4], r0 = P[5], c0 = P[6], Rs = P[7];

    float res = 0.0f;
    const bool inside = (R >= r0) && (R < r0 + shv) && (C >= c0) && (C < c0 + swv);
    if (inside) {
        float sx = ((float)(C - c0) + 0.5f) * (32.0f / (float)swv) - 0.5f;
        sx = fminf(fmaxf(sx, 0.0f), 31.0f);
        const int   x0 = (int)floorf(sx);
        const int   x1 = (x0 + 1) < 31 ? (x0 + 1) : 31;
        const float wx = sx - (float)x0;
        const float* Trow = T + ((size_t)(o * 512 + (R - Rs))) * 32;
        const float local = (1.0f - wx) * Trow[x0] + wx * Trow[x1];
        res = sigmoidf_(local) * sigmoidf_(feat[p]);   // feat channel 0 = saliency
    }
    out[2 * kHW + kNORG * 2 + (size_t)o * kHW + p] = res;
}

// ---------------------------------------------------------------------------
// K5: size map output: abs(trunc_to_int(features[1025:1027]))
// ---------------------------------------------------------------------------
__global__ void k_size(const float* __restrict__ feat, float* __restrict__ out) {
    const int i = blockIdx.x * blockDim.x + threadIdx.x;
    if (i < 2 * kHW) {
        int v = (int)feat[(size_t)1025 * kHW + i];
        out[i] = (float)(v < 0 ? -v : v);
    }
}

// ---------------------------------------------------------------------------
extern "C" void kernel_launch(void* const* d_in, const int* in_sizes, int n_in,
                              void* d_out, int out_size, void* d_ws, size_t ws_size,
                              hipStream_t stream) {
    const float* feat = (const float*)d_in[0];     // (1,1037,512,512) f32
    float* out = (float*)d_out;                    // size(2*HW) | centers(20) | final(10*HW)
    char*  ws  = (char*)d_ws;

    // workspace layout (all 16B-aligned by construction)
    float* pval   = (float*)ws;                                        // NORG*NBLK f32
    int*   pidx   = (int*)(ws + (size_t)kNORG * kNBLK * 4);            // NORG*NBLK i32
    int*   params = (int*)(ws + (size_t)kNORG * kNBLK * 8);            // NORG*16  i32
    float* shape  = (float*)(ws + (size_t)kNORG * kNBLK * 8 + kNORG * 16 * 4); // NORG*1024 f32
    float* T      = shape + (size_t)kNORG * 1024;                      // NORG*512*32 f32

    k_argmax_partial<<<dim3(kNBLK, kNORG), 256, 0, stream>>>(feat, pval, pidx);
    k_argmax_gather <<<kNORG, 1024, 0, stream>>>(feat, pval, pidx, params, shape, out);
    k_rowinterp_wmma<<<dim3(32, kNORG), 32, 0, stream>>>(shape, params, T);
    k_final         <<<dim3(kHW / 256, kNORG), 256, 0, stream>>>(feat, params, T, out);
    k_size          <<<(2 * kHW + 255) / 256, 256, 0, stream>>>(feat, out);
}
